// GINModel_73521250173225
// MI455X (gfx1250) — matrix operand order repaired and checked by reading, not compiled, not run on hardware
//
#include <hip/hip_runtime.h>
#include <hip/hip_bf16.h>

// ---------------------------------------------------------------------------
// GIN (3 layers) + FC head for MI455X (gfx1250, wave32, WMMA).
//   - Aggregation: coalesced float4 gathers + global_atomic_add_f32 scatters
//   - GEMMs: pure-bf16 operand stream into v_wmma_f32_16x16x32_bf16
//     (all f32->bf16 conversion hoisted out of the GEMM inner loop)
// ---------------------------------------------------------------------------

typedef __attribute__((ext_vector_type(16))) __bf16 v16bf;
typedef __attribute__((ext_vector_type(8)))  __bf16 v8bf;
typedef __attribute__((ext_vector_type(8)))  float  v8f;

#define N_NODES   50000
#define F_DIM     512
#define H_DIM     512
#define C_DIM     256
#define E_EDGES   800000
#define K_CENTRAL 1024

// ---------------------------------------------------------------------------
// acc[i] = (2 + eps) * h[i]   (vectorized, eps read from device scalar)
// ---------------------------------------------------------------------------
__global__ __launch_bounds__(256) void init_acc_kernel(
    const float* __restrict__ h, const float* __restrict__ eps,
    float* __restrict__ acc, long long n4)
{
    long long t = (long long)blockIdx.x * blockDim.x + threadIdx.x;
    if (t >= n4) return;
    const float c = 2.0f + *eps;
    float4 v = ((const float4*)h)[t];
    v.x *= c; v.y *= c; v.z *= c; v.w *= c;
    ((float4*)acc)[t] = v;
}

// ---------------------------------------------------------------------------
// acc[dst] += h[src] over all edges. One thread = one edge x 4 contiguous
// features. Feature index is the fastest-varying thread coordinate ->
// coalesced gathers; h (100MB) is L2-resident (192MB L2).
// ---------------------------------------------------------------------------
__global__ __launch_bounds__(256) void scatter_add_kernel(
    const float* __restrict__ h, const int* __restrict__ esrc,
    const int* __restrict__ edst, float* __restrict__ acc)
{
    long long t = (long long)blockIdx.x * blockDim.x + threadIdx.x;
    const long long total = (long long)E_EDGES * (F_DIM / 4);
    if (t >= total) return;
    const int e = (int)(t >> 7);        // F_DIM/4 == 128 chunks per edge
    const int c = (int)(t & 127);
    const int s = esrc[e];
    const int d = edst[e];
    float4 v = ((const float4*)(h + (size_t)s * F_DIM))[c];
    float* p = acc + (size_t)d * F_DIM + (size_t)c * 4;
    atomicAdd(p + 0, v.x);
    atomicAdd(p + 1, v.y);
    atomicAdd(p + 2, v.z);
    atomicAdd(p + 3, v.w);
}

// ---------------------------------------------------------------------------
// f32 -> bf16 bulk conversion, 8 elements / thread (16B stores).
// Hoists all v_cvt_pk_bf16_f32 work out of the GEMM inner loop.
// ---------------------------------------------------------------------------
__global__ __launch_bounds__(256) void f32_to_bf16_kernel(
    const float* __restrict__ in, __bf16* __restrict__ out, long long n8)
{
    long long t = (long long)blockIdx.x * blockDim.x + threadIdx.x;
    if (t >= n8) return;
    float4 a = ((const float4*)in)[2 * t + 0];
    float4 b = ((const float4*)in)[2 * t + 1];
    v8bf o;
    o[0] = (__bf16)a.x; o[1] = (__bf16)a.y; o[2] = (__bf16)a.z; o[3] = (__bf16)a.w;
    o[4] = (__bf16)b.x; o[5] = (__bf16)b.y; o[6] = (__bf16)b.z; o[7] = (__bf16)b.w;
    ((v8bf*)out)[t] = o;
}

// ---------------------------------------------------------------------------
// FC gather: out_bf[m, :] = bf16( x3[cidx[m], :] ), shape K_CENTRAL x C_DIM.
// ---------------------------------------------------------------------------
__global__ __launch_bounds__(256) void gather_bf16_kernel(
    const float* __restrict__ x3, const int* __restrict__ cidx,
    __bf16* __restrict__ out)
{
    const int chunks = C_DIM / 8;   // 32
    long long t = (long long)blockIdx.x * blockDim.x + threadIdx.x;
    if (t >= (long long)K_CENTRAL * chunks) return;
    const int m = (int)(t / chunks);
    const int c = (int)(t % chunks);
    const int row = cidx[m];
    const float4* src = (const float4*)(x3 + (size_t)row * C_DIM) + 2 * c;
    float4 a = src[0], b = src[1];
    v8bf o;
    o[0] = (__bf16)a.x; o[1] = (__bf16)a.y; o[2] = (__bf16)a.z; o[3] = (__bf16)a.w;
    o[4] = (__bf16)b.x; o[5] = (__bf16)b.y; o[6] = (__bf16)b.z; o[7] = (__bf16)b.w;
    ((v8bf*)(out + (size_t)m * C_DIM))[c] = o;
}

// ---------------------------------------------------------------------------
// Fragment loaders (bf16 memory -> WMMA fragments), ISA 7.12.2 layouts.
// A 16x32 (MxK): lanes 0-15 K=k0+0..7|k0+16..23, lanes 16-31 K=+8|+24.
// B 32x16 (KxN): lane = column; lanes 0-15 K=k0+0..7|k0+16..23, lanes 16-31 K=+8|+24.
// B comes from W (out x in, row-major) as W^T -> contiguous per-lane reads.
// ---------------------------------------------------------------------------
__device__ __forceinline__ v16bf frag_a_bf(const __bf16* __restrict__ row,
                                           int k0, int laneHi)
{
    const __bf16* p = row + k0 + (laneHi ? 8 : 0);
    v8bf lo = *(const v8bf*)p;          // 16B load
    v8bf hi = *(const v8bf*)(p + 16);   // 16B load
    return __builtin_shufflevector(lo, hi, 0, 1, 2, 3, 4, 5, 6, 7,
                                           8, 9, 10, 11, 12, 13, 14, 15);
}

__device__ __forceinline__ v16bf frag_b_bf(const __bf16* __restrict__ row,
                                           int k0, int laneHi)
{
    const __bf16* p = row + k0 + (laneHi ? 8 : 0);
    v8bf lo = *(const v8bf*)p;
    v8bf hi = *(const v8bf*)(p + 16);
    return __builtin_shufflevector(lo, hi, 0, 1, 2, 3, 4, 5, 6, 7,
                                           8, 9, 10, 11, 12, 13, 14, 15);
}

// ---------------------------------------------------------------------------
// Out[m, n] = act( sum_k A[m, k] * W[n, k] + bias[n] )
// Each wave owns a 32x32 output tile: 2 A frags x 2 B frags -> 4 WMMAs per
// K-step, with zero conversion work in the loop. All control flow is
// wave-uniform so EXEC is all-ones at every WMMA (ISA 7.12 requirement).
// ---------------------------------------------------------------------------
template <int KDIM, int NOUT, bool LRELU>
__global__ __launch_bounds__(256) void gemm_bf16_wmma_kernel(
    const __bf16* __restrict__ A, const __bf16* __restrict__ W,
    const float* __restrict__ bias, float* __restrict__ Out, int M)
{
    const int lane   = threadIdx.x & 31;
    const int wave   = threadIdx.x >> 5;
    const int laneLo = lane & 15;
    const int laneHi = lane >> 4;

    const int ntilesN = NOUT / 32;
    const int mtiles  = (M + 31) / 32;
    const int tile    = blockIdx.x * (blockDim.x >> 5) + wave;
    if (tile >= mtiles * ntilesN) return;          // wave-uniform

    const int tm = tile / ntilesN;
    const int tn = tile % ntilesN;

    const int  row0 = tm * 32 + laneLo;
    const bool doHi = (tm * 32 + 31) < M;          // wave-uniform
    const int  row1 = doHi ? row0 + 16 : row0;

    const __bf16* A0 = A + (size_t)row0 * KDIM;
    const __bf16* A1 = A + (size_t)row1 * KDIM;
    const __bf16* W0 = W + (size_t)(tn * 32 + laneLo) * KDIM;
    const __bf16* W1 = W + (size_t)(tn * 32 + 16 + laneLo) * KDIM;

    v8f acc00 = {}; v8f acc01 = {};
    v8f acc10 = {}; v8f acc11 = {};

#pragma unroll 2
    for (int k0 = 0; k0 < KDIM; k0 += 32) {
        v16bf a0 = frag_a_bf(A0, k0, laneHi);
        v16bf a1 = frag_a_bf(A1, k0, laneHi);
        v16bf b0 = frag_b_bf(W0, k0, laneHi);
        v16bf b1 = frag_b_bf(W1, k0, laneHi);
        acc00 = __builtin_amdgcn_wmma_f32_16x16x32_bf16(
            false, a0, false, b0, (short)0, acc00, false, false);
        acc01 = __builtin_amdgcn_wmma_f32_16x16x32_bf16(
            false, a0, false, b1, (short)0, acc01, false, false);
        acc10 = __builtin_amdgcn_wmma_f32_16x16x32_bf16(
            false, a1, false, b0, (short)0, acc10, false, false);
        acc11 = __builtin_amdgcn_wmma_f32_16x16x32_bf16(
            false, a1, false, b1, (short)0, acc11, false, false);
    }

    // D layout (ISA 7.12.2): VGPR j -> row (j + 8*laneHi), col = lane&15.
    const int   col0 = tn * 32 + laneLo;
    const int   col1 = col0 + 16;
    const float bv0  = bias[col0];
    const float bv1  = bias[col1];

#pragma unroll
    for (int j = 0; j < 8; ++j) {
        const int r = tm * 32 + j + laneHi * 8;
        float v0 = acc00[j] + bv0;
        float v1 = acc01[j] + bv1;
        if (LRELU) {
            v0 = (v0 >= 0.0f) ? v0 : 0.2f * v0;
            v1 = (v1 >= 0.0f) ? v1 : 0.2f * v1;
        }
        Out[(size_t)r * NOUT + col0] = v0;
        Out[(size_t)r * NOUT + col1] = v1;
    }
    if (doHi) {
#pragma unroll
        for (int j = 0; j < 8; ++j) {
            const int r = tm * 32 + 16 + j + laneHi * 8;
            float v0 = acc10[j] + bv0;
            float v1 = acc11[j] + bv1;
            if (LRELU) {
                v0 = (v0 >= 0.0f) ? v0 : 0.2f * v0;
                v1 = (v1 >= 0.0f) ? v1 : 0.2f * v1;
            }
            Out[(size_t)r * NOUT + col0] = v0;
            Out[(size_t)r * NOUT + col1] = v1;
        }
    }
}

// ---------------------------------------------------------------------------
// Host-side launch sequence (stream-only; graph-capture safe).
// ---------------------------------------------------------------------------
extern "C" void kernel_launch(void* const* d_in, const int* in_sizes, int n_in,
                              void* d_out, int out_size, void* d_ws, size_t ws_size,
                              hipStream_t stream)
{
    (void)in_sizes; (void)n_in; (void)out_size; (void)ws_size;

    const float* x    = (const float*)d_in[0];
    const int*   eidx = (const int*)  d_in[1];   // (2, E): row0=src, row1=dst
    const int*   cidx = (const int*)  d_in[2];
    const float* W1   = (const float*)d_in[3];
    const float* b1   = (const float*)d_in[4];
    const float* eps1 = (const float*)d_in[5];
    const float* W2   = (const float*)d_in[6];
    const float* b2   = (const float*)d_in[7];
    const float* eps2 = (const float*)d_in[8];
    const float* W3   = (const float*)d_in[9];
    const float* b3   = (const float*)d_in[10];
    const float* eps3 = (const float*)d_in[11];
    const float* Wfc  = (const float*)d_in[12];
    const float* bfc  = (const float*)d_in[13];

    const int* esrc = eidx;
    const int* edst = eidx + E_EDGES;

    float* x3    = (float*)d_out;                            // N x C
    float* fcout = x3 + (size_t)N_NODES * C_DIM;             // K x F

    // ---- workspace carve-up ----
    char* ws = (char*)d_ws;
    float*  bufA   = (float*)ws;                 ws += (size_t)N_NODES * H_DIM * sizeof(float);   // agg (f32)
    float*  bufB   = (float*)ws;                 ws += (size_t)N_NODES * H_DIM * sizeof(float);   // h   (f32)
    __bf16* accbf  = (__bf16*)ws;                ws += (size_t)N_NODES * H_DIM * sizeof(__bf16);  // agg (bf16)
    __bf16* W1bf   = (__bf16*)ws;                ws += (size_t)H_DIM * F_DIM * sizeof(__bf16);
    __bf16* W2bf   = (__bf16*)ws;                ws += (size_t)H_DIM * H_DIM * sizeof(__bf16);
    __bf16* W3bf   = (__bf16*)ws;                ws += (size_t)C_DIM * H_DIM * sizeof(__bf16);
    __bf16* Wfcbf  = (__bf16*)ws;                ws += (size_t)F_DIM * C_DIM * sizeof(__bf16);
    __bf16* fcAbf  = (__bf16*)ws;                ws += (size_t)K_CENTRAL * C_DIM * sizeof(__bf16);

    const int TB = 256;  // 8 waves per block

    const long long n4 = (long long)N_NODES * F_DIM / 4;
    const dim3 gInit((unsigned)((n4 + TB - 1) / TB));

    const long long n8 = (long long)N_NODES * F_DIM / 8;
    const dim3 gCvt((unsigned)((n8 + TB - 1) / TB));

    const long long scat = (long long)E_EDGES * (F_DIM / 4);
    const dim3 gScat((unsigned)((scat + TB - 1) / TB));

    auto cvt_grid = [TB](long long elems) {
        return dim3((unsigned)((elems / 8 + TB - 1) / TB));
    };
    auto gemm_grid = [](int M, int NOUT) {
        int tiles = ((M + 31) / 32) * (NOUT / 32);
        return dim3((unsigned)((tiles + 7) / 8));
    };

    // ---- one-time weight conversion to bf16 ----
    f32_to_bf16_kernel<<<cvt_grid((long long)H_DIM * F_DIM), TB, 0, stream>>>(W1,  W1bf,  (long long)H_DIM * F_DIM / 8);
    f32_to_bf16_kernel<<<cvt_grid((long long)H_DIM * H_DIM), TB, 0, stream>>>(W2,  W2bf,  (long long)H_DIM * H_DIM / 8);
    f32_to_bf16_kernel<<<cvt_grid((long long)C_DIM * H_DIM), TB, 0, stream>>>(W3,  W3bf,  (long long)C_DIM * H_DIM / 8);
    f32_to_bf16_kernel<<<cvt_grid((long long)F_DIM * C_DIM), TB, 0, stream>>>(Wfc, Wfcbf, (long long)F_DIM * C_DIM / 8);

    // ---- Layer 1: h1 = lrelu( (agg(x) + (2+eps1) x) @ W1^T + b1 ) ----
    init_acc_kernel<<<gInit, TB, 0, stream>>>(x, eps1, bufA, n4);
    scatter_add_kernel<<<gScat, TB, 0, stream>>>(x, esrc, edst, bufA);
    f32_to_bf16_kernel<<<gCvt, TB, 0, stream>>>(bufA, accbf, n8);
    gemm_bf16_wmma_kernel<F_DIM, H_DIM, true>
        <<<gemm_grid(N_NODES, H_DIM), TB, 0, stream>>>(accbf, W1bf, b1, bufB, N_NODES);

    // ---- Layer 2 ----
    init_acc_kernel<<<gInit, TB, 0, stream>>>(bufB, eps2, bufA, n4);
    scatter_add_kernel<<<gScat, TB, 0, stream>>>(bufB, esrc, edst, bufA);
    f32_to_bf16_kernel<<<gCvt, TB, 0, stream>>>(bufA, accbf, n8);
    gemm_bf16_wmma_kernel<H_DIM, H_DIM, true>
        <<<gemm_grid(N_NODES, H_DIM), TB, 0, stream>>>(accbf, W2bf, b2, bufB, N_NODES);

    // ---- Layer 3 -> x3 (d_out region 0) ----
    init_acc_kernel<<<gInit, TB, 0, stream>>>(bufB, eps3, bufA, n4);
    scatter_add_kernel<<<gScat, TB, 0, stream>>>(bufB, esrc, edst, bufA);
    f32_to_bf16_kernel<<<gCvt, TB, 0, stream>>>(bufA, accbf, n8);
    gemm_bf16_wmma_kernel<H_DIM, C_DIM, true>
        <<<gemm_grid(N_NODES, C_DIM), TB, 0, stream>>>(accbf, W3bf, b3, x3, N_NODES);

    // ---- FC head: out = x3[central_idx] @ Wfc^T + bfc ----
    {
        const long long gt = (long long)K_CENTRAL * (C_DIM / 8);
        gather_bf16_kernel<<<dim3((unsigned)((gt + TB - 1) / TB)), TB, 0, stream>>>(x3, cidx, fcAbf);
        gemm_bf16_wmma_kernel<C_DIM, F_DIM, false>
            <<<gemm_grid(K_CENTRAL, F_DIM), TB, 0, stream>>>(fcAbf, Wfcbf, bfc, fcout, K_CENTRAL);
    }
}
